// SSRM_FA_29085518529062
// MI455X (gfx1250) — compile-verified, matrix operand order checked by reading, not run on hardware
//
#include <hip/hip_runtime.h>
#include <hip/hip_bf16.h>
#include <math.h>
#include <stdint.h>

// ---------------------------------------------------------------------------
// SSRM block for MI455X (gfx1250, wave32, WMMA + async global->LDS).
// Shapes: B=4, C=64, H=W=256.  Cs=128, Cr=64.  HEADS=4, HEAD=16, BLOCK=8,
// HALO=1, WIN=10, pooled 128x128.
// ---------------------------------------------------------------------------

#define HW_FULL   65536        // 256*256
#define HW_POOL   16384        // 128*128
#define NPIX      262144       // 4*256*256
#define NPIX_P    65536        // 4*128*128

typedef __attribute__((ext_vector_type(16))) _Float16 v16h;
typedef __attribute__((ext_vector_type(8)))  _Float16 v8h;
typedef __attribute__((ext_vector_type(8)))  float    v8f;
typedef __attribute__((ext_vector_type(2)))  float    v2f;

#if __has_builtin(__builtin_amdgcn_wmma_f32_16x16x4_f32)
#define HAVE_F32WMMA 1
#else
#define HAVE_F32WMMA 0
#endif

#if __has_builtin(__builtin_amdgcn_global_load_async_to_lds_b128) && \
    __has_builtin(__builtin_amdgcn_s_wait_asynccnt)
#define HAVE_ASYNC_LDS 1
#else
#define HAVE_ASYNC_LDS 0
#endif

#if HAVE_ASYNC_LDS
// Parameter types per hipcc diagnostic: b128 variant takes
// (int __vector(4) AS1*, int __vector(4) AS3*, imm offset, imm cpol).
typedef int v4i_vs __attribute__((vector_size(16)));
typedef __attribute__((address_space(1))) v4i_vs glb_v4i;
typedef __attribute__((address_space(3))) v4i_vs lds_v4i;
// Flat shared-memory addresses are {aperture_hi32, lds_offset_lo32}; the low
// 32 bits are the LDS byte address, so an integer cast produces the AS3 ptr.
__device__ __forceinline__ lds_v4i* lds_cast4(void* p) {
  return (lds_v4i*)(unsigned int)(unsigned long long)p;
}
__device__ __forceinline__ glb_v4i* glb_cast4(const void* p) {
  return (glb_v4i*)(unsigned long long)p;
}
#endif

// f16 WMMA operand fragment from an LDS row (32 contiguous K halfs, 16B align).
// Lanes 0-15 take K 0-7 then 16-23; lanes 16-31 take K 8-15 then 24-31.
__device__ __forceinline__ v16h load_frag(const _Float16* rowBase, int hs) {
  v8h lo = *(const v8h*)(rowBase + 8 * hs);
  v8h hi = *(const v8h*)(rowBase + 16 + 8 * hs);
  v16h r;
#pragma unroll
  for (int i = 0; i < 8; ++i) { r[i] = lo[i]; r[i + 8] = hi[i]; }
  return r;
}

__device__ __forceinline__ float gelu_exact(float x) {
  return 0.5f * x * (1.0f + erff(x * 0.70710678118654752f));
}

// ---------------------------------------------------------------------------
// LayerNorm over C=64 per pixel (NCHW, channel stride HW_FULL).
// ---------------------------------------------------------------------------
__global__ __launch_bounds__(256) void ln_kernel(const float* __restrict__ x,
                                                 const float* __restrict__ w,
                                                 const float* __restrict__ bcoef,
                                                 float* __restrict__ y) {
  size_t p  = (size_t)blockIdx.x * 256 + threadIdx.x;   // < NPIX
  int    bI = (int)(p >> 16);
  int    hw = (int)(p & 65535);
  const float* xb = x + (size_t)bI * 64 * HW_FULL + hw;
  float vals[64];
  float s = 0.f;
#pragma unroll
  for (int c = 0; c < 64; ++c) { vals[c] = xb[(size_t)c * HW_FULL]; s += vals[c]; }
  float mu = s * (1.f / 64.f);
  float s2 = 0.f;
#pragma unroll
  for (int c = 0; c < 64; ++c) { float d = vals[c] - mu; s2 += d * d; }
  float inv = rsqrtf(s2 * (1.f / 64.f) + 1e-6f);
  float* yb = y + (size_t)bI * 64 * HW_FULL + hw;
#pragma unroll
  for (int c = 0; c < 64; ++c)
    yb[(size_t)c * HW_FULL] = (vals[c] - mu) * inv * w[c] + bcoef[c];
}

// ---------------------------------------------------------------------------
// 1x1 conv as WMMA GEMM.  M = pixels (tile 64/block), N = Cout (tile 32/block).
// Preferred path: fp32 V_WMMA_F32_16X16X4_F32 with async global->LDS staging
// (no conversion, pure DMA).  8 waves/block, one 16x16 D tile each.
// Epilogue modes: 0: +bias | 1: gelu(aux0)*( +bias) | 2: +bias+aux0
//                 3: +bias+aux0+aux1
// ---------------------------------------------------------------------------
__global__ __launch_bounds__(256) void conv1x1_wmma_kernel(
    const float* __restrict__ X, const float* __restrict__ Wm,
    const float* __restrict__ bias, float* __restrict__ Y,
    const float* __restrict__ aux0, const float* __restrict__ aux1,
    int Cin, int Cout, int HWl, int mode) {
  int tid = threadIdx.x;
  int p0  = blockIdx.x * 64;
  int bI  = p0 / HWl;
  int hw0 = p0 - bI * HWl;
  int n0  = blockIdx.y * 32;
  const float* Xb = X + (size_t)bI * Cin * HWl + hw0;

  int wave  = tid >> 5, lane = tid & 31;
  int mt    = wave & 3, nt = wave >> 2;
  int lan16 = lane & 15, hs = lane >> 4;

  v8f acc = {0.f, 0.f, 0.f, 0.f, 0.f, 0.f, 0.f, 0.f};

#if HAVE_F32WMMA
  // fp32 tiles: A K-major (async-friendly: contiguous pixels both sides).
  __shared__ float Af[32][68] __attribute__((aligned(16)));  // k x pixels(+pad)
  __shared__ float Bw[32][36] __attribute__((aligned(16)));  // n x k(+pad)

  int lk  = tid >> 3;          // 0..31 : K index for A loader
  int lp  = (tid & 7) * 8;     // pixel group for A loader
  int lnn = tid >> 3;          // 0..31 : n index for B loader
  int lq  = (tid & 7) * 4;     // K group for B loader
  int mI  = mt * 16 + lan16;
  int nI  = nt * 16 + lan16;

  for (int k0 = 0; k0 < Cin; k0 += 32) {
    if (k0) __syncthreads();
    const float* srcA = Xb + (size_t)(k0 + lk) * HWl + lp;
    const float* srcB = Wm + (size_t)(n0 + lnn) * Cin + k0 + lq;
#if HAVE_ASYNC_LDS
    // 64x32 f32 A tile + 32x32 f32 B tile via async DMA (ASYNCcnt-tracked).
    __builtin_amdgcn_global_load_async_to_lds_b128(glb_cast4(srcA),
                                                   lds_cast4(&Af[lk][lp]), 0, 0);
    __builtin_amdgcn_global_load_async_to_lds_b128(glb_cast4(srcA),
                                                   lds_cast4(&Af[lk][lp]), 16, 0);
    __builtin_amdgcn_global_load_async_to_lds_b128(glb_cast4(srcB),
                                                   lds_cast4(&Bw[lnn][lq]), 0, 0);
    if (k0 + 32 < Cin) __builtin_prefetch(srcA + (size_t)32 * HWl, 0, 1);
    __builtin_amdgcn_s_wait_asynccnt(0);
#else
    float4 f0 = *(const float4*)srcA;
    float4 f1 = *(const float4*)(srcA + 4);
    *(float4*)&Af[lk][lp]     = f0;
    *(float4*)&Af[lk][lp + 4] = f1;
    *(float4*)&Bw[lnn][lq]    = *(const float4*)srcB;
    if (k0 + 32 < Cin) __builtin_prefetch(srcA + (size_t)32 * HWl, 0, 1);
#endif
    __syncthreads();
    // f32 16x16x4 fragments: lanes 0-15 hold K0/K1, lanes 16-31 hold K2/K3.
#pragma unroll
    for (int kk = 0; kk < 8; ++kk) {
      int kb = kk * 4 + 2 * hs;
      v2f av;
      av[0] = Af[kb][mI];
      av[1] = Af[kb + 1][mI];
      v2f bv = *(const v2f*)&Bw[nI][kb];
      acc = __builtin_amdgcn_wmma_f32_16x16x4_f32(false, av, false, bv,
                                                  (short)0, acc, false, false);
    }
  }
#else  // fallback: f16 WMMA path (fp32 -> f16 staged through VGPRs)
  __shared__ _Float16 As[64][40] __attribute__((aligned(16)));
  __shared__ _Float16 Bs[32][40] __attribute__((aligned(16)));
  int lk  = tid >> 3;
  int lp  = (tid & 7) * 8;
  int lnn = tid >> 3;
  int lq  = (tid & 7) * 4;
  for (int k0 = 0; k0 < Cin; k0 += 32) {
    if (k0) __syncthreads();
    const float* src = Xb + (size_t)(k0 + lk) * HWl + lp;
    float4 f0 = *(const float4*)(src);
    float4 f1 = *(const float4*)(src + 4);
    As[lp + 0][lk] = (_Float16)f0.x; As[lp + 1][lk] = (_Float16)f0.y;
    As[lp + 2][lk] = (_Float16)f0.z; As[lp + 3][lk] = (_Float16)f0.w;
    As[lp + 4][lk] = (_Float16)f1.x; As[lp + 5][lk] = (_Float16)f1.y;
    As[lp + 6][lk] = (_Float16)f1.z; As[lp + 7][lk] = (_Float16)f1.w;
    float4 wv4 = *(const float4*)(Wm + (size_t)(n0 + lnn) * Cin + k0 + lq);
    Bs[lnn][lq + 0] = (_Float16)wv4.x; Bs[lnn][lq + 1] = (_Float16)wv4.y;
    Bs[lnn][lq + 2] = (_Float16)wv4.z; Bs[lnn][lq + 3] = (_Float16)wv4.w;
    if (k0 + 32 < Cin) __builtin_prefetch(src + (size_t)32 * HWl, 0, 1);
    __syncthreads();
    v16h af = load_frag(&As[mt * 16 + lan16][0], hs);
    v16h bf = load_frag(&Bs[nt * 16 + lan16][0], hs);
    acc = __builtin_amdgcn_wmma_f32_16x16x32_f16(false, af, false, bf,
                                                 (short)0, acc, false, false);
  }
#endif

  // ---- epilogue (D layout: lane n = lane%16, row m = r + 8*(lane/16))
  int    nG    = n0 + nt * 16 + lan16;
  int    hwB   = hw0 + mt * 16 + hs * 8;
  size_t obase = ((size_t)bI * Cout + nG) * HWl + hwB;
  float  bv2   = bias ? bias[nG] : 0.f;
#pragma unroll
  for (int r = 0; r < 8; ++r) {
    size_t idx = obase + r;
    float  val = acc[r] + bv2;
    if (mode == 1) {
      val = gelu_exact(aux0[idx]) * val;
    } else if (mode == 2) {
      val += aux0[idx];
    } else if (mode == 3) {
      val += aux0[idx] + aux1[idx];
    }
    Y[idx] = val;
  }
}

// ---------------------------------------------------------------------------
// mem() depthwise branch: out = dw1x13 + dw13x1 + dw3x3 (+ all biases).
// ---------------------------------------------------------------------------
__global__ __launch_bounds__(256) void dw_mem_kernel(
    const float* __restrict__ t,
    const float* __restrict__ wh, const float* __restrict__ bh,
    const float* __restrict__ wv, const float* __restrict__ bv,
    const float* __restrict__ wl, const float* __restrict__ bl,
    float* __restrict__ out) {
  size_t idx = (size_t)blockIdx.x * 256 + threadIdx.x;  // < 4*64*HW_FULL
  int hw = (int)(idx & 65535);
  int cb = (int)(idx >> 16);
  int c  = cb & 63;
  int h  = hw >> 8, w = hw & 255;
  const float* plane = t + (size_t)cb * HW_FULL;
  float s = bh[c] + bv[c] + bl[c];
#pragma unroll
  for (int j = 0; j < 13; ++j) {
    int x = w - 6 + j;
    if ((unsigned)x < 256u) s += plane[h * 256 + x] * wh[c * 13 + j];
  }
#pragma unroll
  for (int i = 0; i < 13; ++i) {
    int y = h - 6 + i;
    if ((unsigned)y < 256u) s += plane[y * 256 + w] * wv[c * 13 + i];
  }
#pragma unroll
  for (int i = 0; i < 3; ++i)
#pragma unroll
    for (int j = 0; j < 3; ++j) {
      int y = h - 1 + i, x = w - 1 + j;
      if ((unsigned)y < 256u && (unsigned)x < 256u)
        s += plane[y * 256 + x] * wl[c * 9 + i * 3 + j];
    }
  out[idx] = s;
}

// ---------------------------------------------------------------------------
// 2x2 max-pool (stride 2): 256x256 -> 128x128.
// ---------------------------------------------------------------------------
__global__ __launch_bounds__(256) void maxpool_kernel(const float* __restrict__ v,
                                                      float* __restrict__ out) {
  size_t idx = (size_t)blockIdx.x * 256 + threadIdx.x;  // < 4*64*HW_POOL
  int hw = (int)(idx & 16383);
  int cb = (int)(idx >> 14);
  int y = hw >> 7, x = hw & 127;
  const float* plane = v + (size_t)cb * HW_FULL;
  const float* r0 = plane + (y * 2) * 256 + x * 2;
  float m = fmaxf(fmaxf(r0[0], r0[1]), fmaxf(r0[256], r0[257]));
  out[idx] = m;
}

// ---------------------------------------------------------------------------
// Windowed halo attention.  One block per (window, head, batch).
// Q: 64x16 (scaled), K/V halo window: 100x16 (K gets rel_h/rel_w bias).
// sim = Q K^T (f16 WMMA, K padded to 32), softmax in LDS, O = P V (K=128).
// ---------------------------------------------------------------------------
__global__ __launch_bounds__(256) void attn_kernel(const float* __restrict__ qkv,
                                                   const float* __restrict__ relh,
                                                   const float* __restrict__ relw,
                                                   float* __restrict__ o) {
  __shared__ _Float16 Qs[64][40]   __attribute__((aligned(16)));  // q x ch(pad32)
  __shared__ _Float16 Ks[112][40]  __attribute__((aligned(16)));  // key x ch(pad32)
  __shared__ _Float16 Vs[16][136]  __attribute__((aligned(16)));  // ch x key(pad128)
  __shared__ _Float16 Ps[64][136]  __attribute__((aligned(16)));  // q x key(pad128)
  __shared__ float    Sim[64][113];

  int tid  = threadIdx.x;
  int b    = blockIdx.z;
  int head = blockIdx.y;
  int wy   = blockIdx.x >> 4;
  int wx   = blockIdx.x & 15;

  const float* qbase = qkv + ((size_t)(b * 192 + head * 16)) * HW_POOL;
  const float* kbase = qkv + ((size_t)(b * 192 + 64 + head * 16)) * HW_POOL;
  const float* vbase = qkv + ((size_t)(b * 192 + 128 + head * 16)) * HW_POOL;

  // stage Q (64 queries x 16 ch, scaled by HEAD^-0.5 = 0.25); zero the K-pad.
  for (int e = tid; e < 1024; e += 256) {
    int q = e >> 4, c = e & 15;
    int gh = wy * 8 + (q >> 3), gw = wx * 8 + (q & 7);
    Qs[q][c]      = (_Float16)(qbase[(size_t)c * HW_POOL + gh * 128 + gw] * 0.25f);
    Qs[q][16 + c] = (_Float16)0.0f;
  }
  // stage K (with rel bias, zero-padded halo) and V.
  for (int i = tid; i < 128 * 32; i += 256) {
    int kk = i >> 5, c = i & 31;
    float kv = 0.f, vv = 0.f;
    if (kk < 100 && c < 16) {
      int ki = wy * 8 - 1 + kk / 10;
      int kj = wx * 8 - 1 + kk % 10;
      if (ki >= 0 && ki < 128 && kj >= 0 && kj < 128) {
        size_t off = (size_t)c * HW_POOL + ki * 128 + kj;
        kv = kbase[off];
        vv = vbase[off];
      }
      kv += (c < 8) ? relh[(kk / 10) * 8 + c] : relw[(kk % 10) * 8 + (c - 8)];
    }
    if (kk < 112) Ks[kk][c] = (_Float16)kv;
    if (c < 16)   Vs[c][kk] = (_Float16)vv;
  }
  __syncthreads();

  int wave = tid >> 5, lane = tid & 31;
  int lan16 = lane & 15, hs = lane >> 4;

  // sim = Q K^T : 4 M-tiles x 7 N-tiles, one 16x16x32 WMMA each.
  for (int t = wave; t < 28; t += 8) {
    int mt = t & 3, nt = t >> 2;
    v16h af = load_frag(&Qs[mt * 16 + lan16][0], hs);
    v16h bf = load_frag(&Ks[nt * 16 + lan16][0], hs);
    v8f acc = {0.f, 0.f, 0.f, 0.f, 0.f, 0.f, 0.f, 0.f};
    acc = __builtin_amdgcn_wmma_f32_16x16x32_f16(false, af, false, bf,
                                                 (short)0, acc, false, false);
#pragma unroll
    for (int r = 0; r < 8; ++r)
      Sim[mt * 16 + hs * 8 + r][nt * 16 + lan16] = acc[r];
  }
  __syncthreads();

  // softmax over the 100 real keys; write normalized probs (f16, pad to 128).
  if (tid < 64) {
    float mx = -1e30f;
    for (int j = 0; j < 100; ++j) mx = fmaxf(mx, Sim[tid][j]);
    float s = 0.f;
    for (int j = 0; j < 100; ++j) {
      float e = __expf(Sim[tid][j] - mx);
      s += e;
      Sim[tid][j] = e;
    }
    float inv = 1.f / s;
    for (int j = 0; j < 100; ++j) Ps[tid][j] = (_Float16)(Sim[tid][j] * inv);
    for (int j = 100; j < 128; ++j) Ps[tid][j] = (_Float16)0.0f;
  }
  __syncthreads();

  // O = P V : 4 M-tiles, N=16, K=128 (4 WMMA accumulations).  Waves 0-3.
  if (wave < 4) {
    int mt = wave;
    v8f acc = {0.f, 0.f, 0.f, 0.f, 0.f, 0.f, 0.f, 0.f};
#pragma unroll
    for (int kc = 0; kc < 4; ++kc) {
      v16h af = load_frag(&Ps[mt * 16 + lan16][kc * 32], hs);
      v16h bf = load_frag(&Vs[lan16][kc * 32], hs);
      acc = __builtin_amdgcn_wmma_f32_16x16x32_f16(false, af, false, bf,
                                                   (short)0, acc, false, false);
    }
    int q0 = mt * 16 + hs * 8;
    int gh = wy * 8 + (q0 >> 3);
    size_t base = ((size_t)(b * 64 + head * 16 + lan16)) * HW_POOL + gh * 128 + wx * 8;
#pragma unroll
    for (int r = 0; r < 8; ++r) o[base + r] = acc[r];
  }
}

// ---------------------------------------------------------------------------
// 3x3 depthwise conv with reflect padding (pooled res).
// ---------------------------------------------------------------------------
__global__ __launch_bounds__(256) void refl_conv_kernel(const float* __restrict__ x,
                                                        const float* __restrict__ wl,
                                                        const float* __restrict__ bl,
                                                        float* __restrict__ out) {
  size_t idx = (size_t)blockIdx.x * 256 + threadIdx.x;  // < 4*64*HW_POOL
  int hw = (int)(idx & 16383);
  int cb = (int)(idx >> 14);
  int c  = cb & 63;
  int y  = hw >> 7, xx = hw & 127;
  const float* plane = x + (size_t)cb * HW_POOL;
  float s = bl[c];
#pragma unroll
  for (int i = 0; i < 3; ++i)
#pragma unroll
    for (int j = 0; j < 3; ++j) {
      int yy = y - 1 + i;
      int xj = xx - 1 + j;
      yy = (yy < 0) ? -yy : ((yy > 127) ? 254 - yy : yy);
      xj = (xj < 0) ? -xj : ((xj > 127) ? 254 - xj : xj);
      s += plane[yy * 128 + xj] * wl[c * 9 + i * 3 + j];
    }
  out[idx] = s;
}

// ---------------------------------------------------------------------------
// Bilinear x2 upsample (align-corners linspace, 127/255 scale) with fused
// residual epilogue: out = up(x) + v - bb.
// ---------------------------------------------------------------------------
__global__ __launch_bounds__(256) void upsample_res_kernel(
    const float* __restrict__ x, const float* __restrict__ v,
    const float* __restrict__ bb, float* __restrict__ out) {
  size_t idx = (size_t)blockIdx.x * 256 + threadIdx.x;  // < 4*64*HW_FULL
  int hw = (int)(idx & 65535);
  int cb = (int)(idx >> 16);
  int oy = hw >> 8, ox = hw & 255;
  const float scale = 127.0f / 255.0f;
  float sy = oy * scale, sx = ox * scale;
  int y0 = (int)sy, x0 = (int)sx;
  float fy = sy - y0, fx = sx - x0;
  int y1 = (y0 + 1 < 128) ? y0 + 1 : 127;
  int x1 = (x0 + 1 < 128) ? x0 + 1 : 127;
  const float* plane = x + (size_t)cb * HW_POOL;
  float tl = plane[y0 * 128 + x0], tr = plane[y0 * 128 + x1];
  float blv = plane[y1 * 128 + x0], br = plane[y1 * 128 + x1];
  float top = tl + (tr - tl) * fx;
  float bot = blv + (br - blv) * fx;
  float val = top + (bot - top) * fy;
  out[idx] = val + v[idx] - bb[idx];
}

// ---------------------------------------------------------------------------
// Host side
// ---------------------------------------------------------------------------
struct MemP {
  const float *in1_w, *in1_b, *pw1_w, *pw1_b, *lkh_w, *lkh_b, *lkv_w, *lkv_b;
  const float *loc_w, *loc_b, *pw2_w, *pw2_b, *out_w, *out_b;
};

static MemP get_mem(void* const* d, int o) {
  MemP p;
  p.in1_w = (const float*)d[o + 0];  p.in1_b = (const float*)d[o + 1];
  p.pw1_w = (const float*)d[o + 2];  p.pw1_b = (const float*)d[o + 3];
  p.lkh_w = (const float*)d[o + 4];  p.lkh_b = (const float*)d[o + 5];
  p.lkv_w = (const float*)d[o + 6];  p.lkv_b = (const float*)d[o + 7];
  p.loc_w = (const float*)d[o + 8];  p.loc_b = (const float*)d[o + 9];
  p.pw2_w = (const float*)d[o + 10]; p.pw2_b = (const float*)d[o + 11];
  p.out_w = (const float*)d[o + 12]; p.out_b = (const float*)d[o + 13];
  return p;
}

static void gemm1x1(const float* X, const float* Wm, const float* bias, float* Y,
                    int Cin, int Cout, int HWl, int NPl, int mode,
                    const float* aux0, const float* aux1, hipStream_t s) {
  dim3 grid(NPl / 64, Cout / 32);
  conv1x1_wmma_kernel<<<grid, 256, 0, s>>>(X, Wm, bias, Y, aux0, aux1,
                                           Cin, Cout, HWl, mode);
}

// mem(x, p): x1 = in1(x); t = pw1(x); x2 = dw(t); gated = gelu(x1)*pw2(x2);
// out = out_w(gated) [+ epilogue residuals]
static void run_mem(const float* x, const MemP& p, float* x1b, float* tb,
                    float* x2b, float* outb, int outMode, const float* aux0,
                    const float* aux1, hipStream_t s) {
  gemm1x1(x, p.in1_w, p.in1_b, x1b, 64, 128, HW_FULL, NPIX, 0, nullptr, nullptr, s);
  gemm1x1(x, p.pw1_w, p.pw1_b, tb, 64, 64, HW_FULL, NPIX, 0, nullptr, nullptr, s);
  dw_mem_kernel<<<NPIX * 64 / 256, 256, 0, s>>>(tb, p.lkh_w, p.lkh_b, p.lkv_w,
                                                p.lkv_b, p.loc_w, p.loc_b, x2b);
  gemm1x1(x2b, p.pw2_w, p.pw2_b, x1b, 64, 128, HW_FULL, NPIX, 1, x1b, nullptr, s);
  gemm1x1(x1b, p.out_w, p.out_b, outb, 128, 64, HW_FULL, NPIX, outMode, aux0, aux1, s);
}

extern "C" void kernel_launch(void* const* d_in, const int* in_sizes, int n_in,
                              void* d_out, int out_size, void* d_ws, size_t ws_size,
                              hipStream_t stream) {
  (void)in_sizes; (void)n_in; (void)out_size; (void)ws_size;
  const float* a    = (const float*)d_in[0];
  const float* y    = (const float*)d_in[1];
  const float* ln_w = (const float*)d_in[2];
  const float* ln_b = (const float*)d_in[3];
  MemP s1  = get_mem(d_in, 4);
  MemP s2  = get_mem(d_in, 18);
  const float* qkv_w = (const float*)d_in[32];
  const float* rel_h = (const float*)d_in[33];
  const float* rel_w = (const float*)d_in[34];
  const float* lp_w  = (const float*)d_in[35];
  const float* lp_b  = (const float*)d_in[36];
  MemP s4  = get_mem(d_in, 37);
  MemP ffn = get_mem(d_in, 51);

  float* ws = (float*)d_ws;
  const size_t NP = NPIX;
  float* bufA  = ws;                   // NP*64  : a_ln, later a2 (pooled-v alias)
  float* bufX1 = bufA + NP * 64;       // NP*128 : x1 / gated (pooled-qkv alias)
  float* bufT  = bufX1 + NP * 128;     // NP*64  : t (pooled attn-out alias)
  float* bufX2 = bufT + NP * 64;       // NP*64  : x2 (pooled lp-conv alias)
  float* bufV  = bufX2 + NP * 64;      // NP*64  : v, later a2b
  float* bufBB = bufV + NP * 64;       // NP*64  : bb
  float* bufP   = bufA;                // pooled v (dead before bufA = a2)
  float* bufQKV = bufX1;               // pooled qkv (192ch <= 128ch full-res)
  float* bufO   = bufT;                // pooled attn output
  float* bufO2  = bufX2;               // pooled lp-conv output
  float* out    = (float*)d_out;

  // 1. a_ln = LayerNorm(a)
  ln_kernel<<<NPIX / 256, 256, 0, stream>>>(a, ln_w, ln_b, bufA);
  // 2. v = mem(a_ln, s1) + a_ln + y          (fused residual epilogue)
  run_mem(bufA, s1, bufX1, bufT, bufX2, bufV, 3, bufA, y, stream);
  // 3. bb = mem(a_ln, s2)
  run_mem(bufA, s2, bufX1, bufT, bufX2, bufBB, 0, nullptr, nullptr, stream);
  // 4. xp = maxpool2x2(v)
  maxpool_kernel<<<NPIX_P * 64 / 256, 256, 0, stream>>>(bufV, bufP);
  // 5. qkv = conv1x1(xp)  (no bias)
  gemm1x1(bufP, qkv_w, nullptr, bufQKV, 64, 192, HW_POOL, NPIX_P, 0,
          nullptr, nullptr, stream);
  // 6. windowed halo attention
  attn_kernel<<<dim3(256, 4, 4), 256, 0, stream>>>(bufQKV, rel_h, rel_w, bufO);
  // 7. reflect-pad 3x3 depthwise conv
  refl_conv_kernel<<<NPIX_P * 64 / 256, 256, 0, stream>>>(bufO, lp_w, lp_b, bufO2);
  // 8. a2 = upsample(o2) + v - bb            (fused epilogue)
  upsample_res_kernel<<<NPIX * 64 / 256, 256, 0, stream>>>(bufO2, bufV, bufBB, bufA);
  // 9. a2b = mem(a2, s4) + bb
  run_mem(bufA, s4, bufX1, bufT, bufX2, bufV, 2, bufBB, nullptr, stream);
  // 10. out = mem(a2b, ffn) + a2b
  run_mem(bufV, ffn, bufX1, bufT, bufX2, out, 2, bufV, nullptr, stream);
}
